// GraphEncoder_67078799229098
// MI455X (gfx1250) — compile-verified, matrix operand order checked by reading, not run on hardware
//
#include <hip/hip_runtime.h>

#define D_MODEL 768
#define NLAYERS 4

typedef __attribute__((ext_vector_type(16))) __bf16 v16bf;
typedef __attribute__((ext_vector_type(8)))  __bf16 v8bf;
typedef __attribute__((ext_vector_type(8)))  float  v8f;

enum { ACT_NONE = 0, ACT_RELU = 1, ACT_GELU = 2 };

__device__ __forceinline__ float gelu_f(float x) {
    const float c0 = 0.7978845608028654f;   // sqrt(2/pi)
    const float c1 = 0.044715f;
    return 0.5f * x * (1.0f + tanhf(c0 * (x + c1 * x * x * x)));
}

// ---- CDNA5 async copy: global -> LDS, 16B per lane, tracked by ASYNCcnt ----
__device__ __forceinline__ void async_copy_b128(unsigned lds_addr, const void* gptr) {
    asm volatile("global_load_async_to_lds_b128 %0, %1, off"
                 :: "v"(lds_addr), "v"((unsigned long long)gptr)
                 : "memory");
}
__device__ __forceinline__ void async_wait_all() {
    asm volatile("s_wait_asynccnt 0x0" ::: "memory");
}
// Address-translation fence so the compiler may reuse address VGPRs of the
// inline-asm async loads (XCNT is invisible to compiler bookkeeping here).
__device__ __forceinline__ void xcnt_fence() {
    asm volatile("s_wait_xcnt 0x0" ::: "memory");
}

// -------- block-wide sum + sumsq reduction (256 threads = 8 wave32) --------
__device__ __forceinline__ void block_reduce_2(float& s, float& q) {
    #pragma unroll
    for (int off = 16; off > 0; off >>= 1) {
        s += __shfl_down(s, off);
        q += __shfl_down(q, off);
    }
    __shared__ float rs[8];
    __shared__ float rq[8];
    const int lane = threadIdx.x & 31;
    const int wv   = threadIdx.x >> 5;
    if (lane == 0) { rs[wv] = s; rq[wv] = q; }
    __syncthreads();
    if (wv == 0) {
        s = (lane < 8) ? rs[lane] : 0.0f;
        q = (lane < 8) ? rq[lane] : 0.0f;
        #pragma unroll
        for (int off = 4; off > 0; off >>= 1) {
            s += __shfl_down(s, off);
            q += __shfl_down(q, off);
        }
        if (lane == 0) { rs[0] = s; rq[0] = q; }
    }
    __syncthreads();
    s = rs[0];
    q = rq[0];
}

// -------- weight fp32 [K][N] -> bf16 transposed [N][K] (768x768) --------
__global__ __launch_bounds__(256) void k_transpose_bf16(
    const float* __restrict__ W, __bf16* __restrict__ Wt)
{
    __shared__ float tile[16][17];
    const int bx = blockIdx.x * 16;   // n base
    const int by = blockIdx.y * 16;   // k base
    const int tx = threadIdx.x;       // 0..15
    const int ty = threadIdx.y;       // 0..15
    tile[ty][tx] = W[(size_t)(by + ty) * D_MODEL + (bx + tx)];
    __syncthreads();
    Wt[(size_t)(bx + ty) * D_MODEL + (by + tx)] = (__bf16)tile[tx][ty];
}

// -------- embedding sum + LayerNorm -> bf16 activations --------
__global__ __launch_bounds__(256) void k_embed_ln(
    const int* __restrict__ ids, const float* __restrict__ emb,
    const float* __restrict__ g, const float* __restrict__ bi,
    __bf16* __restrict__ out, int F, int V)
{
    const int row = blockIdx.x;
    const int tid = threadIdx.x;
    float vals[3];
    float s = 0.0f, q = 0.0f;
    #pragma unroll
    for (int j = 0; j < 3; ++j) {
        const int d = tid + j * 256;
        float a = 0.0f;
        for (int f = 0; f < F; ++f) {
            const int id = ids[row * F + f];
            a += emb[((size_t)f * V + id) * D_MODEL + d];
        }
        vals[j] = a;
        s += a;
        q += a * a;
    }
    block_reduce_2(s, q);
    const float mean = s * (1.0f / D_MODEL);
    const float var  = q * (1.0f / D_MODEL) - mean * mean;
    const float rstd = rsqrtf(var + 1e-5f);
    #pragma unroll
    for (int j = 0; j < 3; ++j) {
        const int d = tid + j * 256;
        out[(size_t)row * D_MODEL + d] =
            (__bf16)((vals[j] - mean) * rstd * g[d] + bi[d]);
    }
}

// -------- WMMA bf16 GEMM: out = act(A[M x 768] * W[768 x 768] + bias) --------
// Wt is stored transposed+bf16: Wt[n][k]. Double-buffered LDS staged via
// global_load_async_to_lds_b128; compute overlaps next tile's fill.
#define BM 128
#define BN 64
#define BK 32
#define LDT (BK + 8)   // halfword stride; 80B rows -> conflict-free b128 LDS reads

template<int ACT, bool OUT_BF16>
__global__ __launch_bounds__(256) void k_gemm_wmma(
    const __bf16* __restrict__ A, const __bf16* __restrict__ Wt,
    const float* __restrict__ bias, void* __restrict__ outv, int Mrows)
{
    __shared__ __align__(16) __bf16 As[2][BM][LDT];
    __shared__ __align__(16) __bf16 Bs[2][BN][LDT];

    const int tid  = threadIdx.x;
    const int lane = tid & 31;
    const int wave = tid >> 5;          // 0..7
    const int m0   = (wave & 3) * 32;   // wave M offset within block tile
    const int n0   = (wave >> 2) * 32;  // wave N offset within block tile
    const int blockM = blockIdx.y * BM;
    const int blockN = blockIdx.x * BN;

    // 16-bit A-matrix lane layout (ISA 7.12.2): lanes 0-15 hold K 0-7 / 16-23,
    // lanes 16-31 hold K 8-15 / 24-31; row M = lane % 16.
    const int frow = lane & 15;
    const int koff = (lane < 16) ? 0 : 8;

    // ---- staging thread map: thread t covers A chunks (r, kb) and (r+64, kb),
    //      plus B chunk (rb, kb); all 16B, clamped rows (masked at epilogue).
    const int r  = tid >> 2;            // 0..63
    const int kb = (tid & 3) << 3;      // 0,8,16,24
    const int gr0 = min(blockM + r,      Mrows - 1);
    const int gr1 = min(blockM + r + 64, Mrows - 1);
    const int gb  = blockN + r;

    const unsigned asz = (unsigned)sizeof(As[0]);
    const unsigned bsz = (unsigned)sizeof(Bs[0]);
    const unsigned ldsA0 = (unsigned)(unsigned long long)&As[0][r][kb];
    const unsigned ldsA1 = (unsigned)(unsigned long long)&As[0][r + 64][kb];
    const unsigned ldsB  = (unsigned)(unsigned long long)&Bs[0][r][kb];

    const __bf16* pa0 = A  + (size_t)gr0 * D_MODEL + kb;
    const __bf16* pa1 = A  + (size_t)gr1 * D_MODEL + kb;
    const __bf16* pb  = Wt + (size_t)gb  * D_MODEL + kb;

    auto issue = [&](int k0, int buf) {
        async_copy_b128(ldsA0 + buf * asz, pa0 + k0);
        async_copy_b128(ldsA1 + buf * asz, pa1 + k0);
        async_copy_b128(ldsB  + buf * bsz, pb  + k0);
        xcnt_fence();
    };

    v8f acc[2][2] = {};

    issue(0, 0);
    int cur = 0;
    for (int k0 = 0; k0 < D_MODEL; k0 += BK) {
        async_wait_all();       // this wave's fills of buffer `cur` are done
        __syncthreads();        // ... and everyone else's; prior readers drained
        if (k0 + BK < D_MODEL)
            issue(k0 + BK, cur ^ 1);   // prefetch next tile into other buffer

        v16bf af[2], bfr[2];
        #pragma unroll
        for (int t = 0; t < 2; ++t) {
            const int rr = m0 + t * 16 + frow;
            v8bf lo = *reinterpret_cast<const v8bf*>(&As[cur][rr][koff]);
            v8bf hi = *reinterpret_cast<const v8bf*>(&As[cur][rr][16 + koff]);
            af[t] = __builtin_shufflevector(lo, hi,
                0, 1, 2, 3, 4, 5, 6, 7, 8, 9, 10, 11, 12, 13, 14, 15);
        }
        #pragma unroll
        for (int u = 0; u < 2; ++u) {
            const int rr = n0 + u * 16 + frow;
            v8bf lo = *reinterpret_cast<const v8bf*>(&Bs[cur][rr][koff]);
            v8bf hi = *reinterpret_cast<const v8bf*>(&Bs[cur][rr][16 + koff]);
            bfr[u] = __builtin_shufflevector(lo, hi,
                0, 1, 2, 3, 4, 5, 6, 7, 8, 9, 10, 11, 12, 13, 14, 15);
        }

        #pragma unroll
        for (int t = 0; t < 2; ++t)
            #pragma unroll
            for (int u = 0; u < 2; ++u)
                acc[t][u] = __builtin_amdgcn_wmma_f32_16x16x32_bf16(
                    false, af[t], false, bfr[u],
                    (short)0, acc[t][u], false, false);
        cur ^= 1;
    }

    // Epilogue. C/D layout: lanes 0-15 -> M=v, N=lane; lanes 16-31 -> M=v+8.
    const int mhalf = (lane >= 16) ? 8 : 0;
    const int nb    = lane & 15;
    #pragma unroll
    for (int t = 0; t < 2; ++t) {
        #pragma unroll
        for (int u = 0; u < 2; ++u) {
            const int col = blockN + n0 + u * 16 + nb;
            const float bv = bias[col];
            #pragma unroll
            for (int v = 0; v < 8; ++v) {
                const int row = blockM + m0 + t * 16 + mhalf + v;
                if (row < Mrows) {
                    float x = acc[t][u][v] + bv;
                    if (ACT == ACT_RELU) x = fmaxf(x, 0.0f);
                    if (ACT == ACT_GELU) x = gelu_f(x);
                    if (OUT_BF16)
                        reinterpret_cast<__bf16*>(outv)[(size_t)row * D_MODEL + col] = (__bf16)x;
                    else
                        reinterpret_cast<float*>(outv)[(size_t)row * D_MODEL + col] = x;
                }
            }
        }
    }
}

// -------- zero fp32 buffer --------
__global__ __launch_bounds__(256) void k_zero(float* __restrict__ p, long n) {
    const long i = (long)blockIdx.x * 256 + threadIdx.x;
    if (i < n) p[i] = 0.0f;
}

// -------- edge messages: agg[dst] += relu(h[src] + e) --------
__global__ __launch_bounds__(192) void k_message(
    const float* __restrict__ h, const float* __restrict__ e,
    const int* __restrict__ src, const int* __restrict__ dst,
    float* __restrict__ agg)
{
    const int eid = blockIdx.x;
    const int c   = threadIdx.x * 4;
    const int s   = src[eid];
    const int d   = dst[eid];
    const float4 hv = *reinterpret_cast<const float4*>(h + (size_t)s * D_MODEL + c);
    const float4 ev = *reinterpret_cast<const float4*>(e + (size_t)eid * D_MODEL + c);
    float4 m;
    m.x = fmaxf(hv.x + ev.x, 0.0f);
    m.y = fmaxf(hv.y + ev.y, 0.0f);
    m.z = fmaxf(hv.z + ev.z, 0.0f);
    m.w = fmaxf(hv.w + ev.w, 0.0f);
    float* ap = agg + (size_t)d * D_MODEL + c;
    __hip_atomic_fetch_add(ap + 0, m.x, __ATOMIC_RELAXED, __HIP_MEMORY_SCOPE_AGENT);
    __hip_atomic_fetch_add(ap + 1, m.y, __ATOMIC_RELAXED, __HIP_MEMORY_SCOPE_AGENT);
    __hip_atomic_fetch_add(ap + 2, m.z, __ATOMIC_RELAXED, __HIP_MEMORY_SCOPE_AGENT);
    __hip_atomic_fetch_add(ap + 3, m.w, __ATOMIC_RELAXED, __HIP_MEMORY_SCOPE_AGENT);
}

// -------- z = bf16(h + agg) (GEMM input prep) --------
__global__ __launch_bounds__(256) void k_zprep(
    const float* __restrict__ h, const float* __restrict__ agg,
    __bf16* __restrict__ out, long n)
{
    const long i = (long)blockIdx.x * 256 + threadIdx.x;
    if (i < n) out[i] = (__bf16)(h[i] + agg[i]);
}

// -------- h = LayerNorm(gelu(z) + h_in) --------
__global__ __launch_bounds__(256) void k_gelu_res_ln(
    const float* __restrict__ z, const float* __restrict__ hin,
    const float* __restrict__ g, const float* __restrict__ bi,
    float* __restrict__ out)
{
    const int row = blockIdx.x;
    const int tid = threadIdx.x;
    float vals[3];
    float s = 0.0f, q = 0.0f;
    #pragma unroll
    for (int j = 0; j < 3; ++j) {
        const int d = tid + j * 256;
        const size_t idx = (size_t)row * D_MODEL + d;
        const float t = gelu_f(z[idx]) + hin[idx];
        vals[j] = t;
        s += t;
        q += t * t;
    }
    block_reduce_2(s, q);
    const float mean = s * (1.0f / D_MODEL);
    const float var  = q * (1.0f / D_MODEL) - mean * mean;
    const float rstd = rsqrtf(var + 1e-5f);
    #pragma unroll
    for (int j = 0; j < 3; ++j) {
        const int d = tid + j * 256;
        out[(size_t)row * D_MODEL + d] =
            (vals[j] - mean) * rstd * g[d] + bi[d];
    }
}

extern "C" void kernel_launch(void* const* d_in, const int* in_sizes, int n_in,
                              void* d_out, int out_size, void* d_ws, size_t ws_size,
                              hipStream_t stream)
{
    const int*   x          = (const int*)  d_in[0];
    const int*   edge_attr  = (const int*)  d_in[1];
    const int*   edge_index = (const int*)  d_in[2];
    const float* atom_emb   = (const float*)d_in[3];
    const float* atom_ln_g  = (const float*)d_in[4];
    const float* atom_ln_b  = (const float*)d_in[5];
    const float* atom_w1    = (const float*)d_in[6];
    const float* atom_b1    = (const float*)d_in[7];
    const float* atom_w2    = (const float*)d_in[8];
    const float* atom_b2    = (const float*)d_in[9];
    const float* bond_emb   = (const float*)d_in[10];
    const float* bond_ln_g  = (const float*)d_in[11];
    const float* bond_ln_b  = (const float*)d_in[12];
    const float* bond_w1    = (const float*)d_in[13];
    const float* bond_b1    = (const float*)d_in[14];
    const float* bond_w2    = (const float*)d_in[15];
    const float* bond_b2    = (const float*)d_in[16];
    const float* conv_w1    = (const float*)d_in[17];
    const float* conv_b1    = (const float*)d_in[18];
    const float* conv_w2    = (const float*)d_in[19];
    const float* conv_b2    = (const float*)d_in[20];
    const float* ln_g       = (const float*)d_in[21];
    const float* ln_b       = (const float*)d_in[22];

    const int N = in_sizes[0] / 9;
    const int E = in_sizes[1] / 3;
    const int* src = edge_index;
    const int* dst = edge_index + E;

    const size_t DD = (size_t)D_MODEL * D_MODEL;

    // ---- carve workspace ----
    char* wsp = (char*)d_ws;
    auto carve = [&](size_t bytes) -> void* {
        void* p = (void*)wsp;
        wsp += (bytes + 255) & ~(size_t)255;
        return p;
    };
    __bf16* wt   = (__bf16*)carve(12 * DD * sizeof(__bf16));
    float*  h    = (float*) carve((size_t)N * D_MODEL * sizeof(float));
    float*  ebuf = (float*) carve((size_t)E * D_MODEL * sizeof(float));
    float*  agg  = (float*) carve((size_t)N * D_MODEL * sizeof(float));
    const size_t maxR = (size_t)((N > E) ? N : E);
    __bf16* bufA = (__bf16*)carve(maxR * D_MODEL * sizeof(__bf16));
    __bf16* bufH = (__bf16*)carve(maxR * D_MODEL * sizeof(__bf16));
    float*  zbuf = (float*) carve((size_t)N * D_MODEL * sizeof(float));

    // ---- 1) convert all weights to bf16, transposed N-major ----
    dim3 tb(16, 16), tg(D_MODEL / 16, D_MODEL / 16);
    k_transpose_bf16<<<tg, tb, 0, stream>>>(atom_w1, wt + 0 * DD);
    k_transpose_bf16<<<tg, tb, 0, stream>>>(atom_w2, wt + 1 * DD);
    k_transpose_bf16<<<tg, tb, 0, stream>>>(bond_w1, wt + 2 * DD);
    k_transpose_bf16<<<tg, tb, 0, stream>>>(bond_w2, wt + 3 * DD);
    for (int l = 0; l < NLAYERS; ++l) {
        k_transpose_bf16<<<tg, tb, 0, stream>>>(conv_w1 + l * DD, wt + (4 + l) * DD);
        k_transpose_bf16<<<tg, tb, 0, stream>>>(conv_w2 + l * DD, wt + (8 + l) * DD);
    }

    const dim3 gN(D_MODEL / BN, (N + BM - 1) / BM);
    const dim3 gE(D_MODEL / BN, (E + BM - 1) / BM);

    // ---- 2) atom encoder: h = (gelu(LN(sum emb) @ w1 + b1)) @ w2 + b2 ----
    k_embed_ln<<<N, 256, 0, stream>>>(x, atom_emb, atom_ln_g, atom_ln_b, bufA, 9, 128);
    k_gemm_wmma<ACT_GELU, true ><<<gN, 256, 0, stream>>>(bufA, wt + 0 * DD, atom_b1, bufH, N);
    k_gemm_wmma<ACT_NONE, false><<<gN, 256, 0, stream>>>(bufH, wt + 1 * DD, atom_b2, h, N);

    // ---- 3) bond encoder -> e ----
    k_embed_ln<<<E, 256, 0, stream>>>(edge_attr, bond_emb, bond_ln_g, bond_ln_b, bufA, 3, 8);
    k_gemm_wmma<ACT_GELU, true ><<<gE, 256, 0, stream>>>(bufA, wt + 2 * DD, bond_b1, bufH, E);
    k_gemm_wmma<ACT_NONE, false><<<gE, 256, 0, stream>>>(bufH, wt + 3 * DD, bond_b2, ebuf, E);

    // ---- 4) GINE layers ----
    const long nElems = (long)N * D_MODEL;
    const int  zgrid  = (int)((nElems + 255) / 256);
    for (int l = 0; l < NLAYERS; ++l) {
        k_zero<<<zgrid, 256, 0, stream>>>(agg, nElems);
        k_message<<<E, D_MODEL / 4, 0, stream>>>(h, ebuf, src, dst, agg);
        k_zprep<<<zgrid, 256, 0, stream>>>(h, agg, bufA, nElems);
        k_gemm_wmma<ACT_RELU, true ><<<gN, 256, 0, stream>>>(
            bufA, wt + (4 + l) * DD, conv_b1 + l * D_MODEL, bufH, N);
        k_gemm_wmma<ACT_NONE, false><<<gN, 256, 0, stream>>>(
            bufH, wt + (8 + l) * DD, conv_b2 + l * D_MODEL, zbuf, N);
        float* outp = (l == NLAYERS - 1) ? (float*)d_out : h;
        k_gelu_res_ln<<<N, 256, 0, stream>>>(
            zbuf, h, ln_g + l * D_MODEL, ln_b + l * D_MODEL, outp);
    }

    (void)n_in; (void)out_size; (void)ws_size;
}
